// GW_12240656793851
// MI455X (gfx1250) — matrix operand order repaired and checked by reading, not compiled, not run on hardware
//
#include <hip/hip_runtime.h>
#include <hip/hip_bf16.h>

typedef __attribute__((ext_vector_type(16))) __bf16 v16bf;
typedef __attribute__((ext_vector_type(8)))  __bf16 v8bf;
typedef __attribute__((ext_vector_type(8)))  float  v8f;

#define NT 256
#define PLANE 65536   // 256*256
#define BATCH 4
#define CH 32

// ---- WMMA fragment loaders (bf16, 16x16x32) ----
// A operand (16x32, row-major source): lane l15 = row; half selects K subset
//   elements 0..7  -> K = half*8 + e ; elements 8..15 -> K = 16 + half*8 + (e-8)
static __device__ __forceinline__ v16bf load_a_frag(const __bf16* rowk, int hf) {
  union { v16bf v; v8bf h[2]; } u;
  u.h[0] = *(const v8bf*)(rowk + hf * 8);
  u.h[1] = *(const v8bf*)(rowk + 16 + hf * 8);
  return u.v;
}
// B operand (32x16): lane l15 = column; elements 0..7 -> K = half*8 + e ;
// elements 8..15 -> K = 16 + half*8 + (e-8) (per-column K data stored K-contiguous)
static __device__ __forceinline__ v16bf load_b_frag(const __bf16* colk, int hf) {
  union { v16bf v; v8bf h[2]; } u;
  const __bf16* p = colk + hf * 8;
  u.h[0] = *(const v8bf*)(p);
  u.h[1] = *(const v8bf*)(p + 16);
  return u.v;
}

// ---- adjacency prep: transpose supports to bf16 ----
__global__ void transpose_bf16(const float* __restrict__ S, __bf16* __restrict__ At) {
  const int j = blockIdx.x * 256 + threadIdx.x;     // 65536 threads
  const int v = j >> 8, w = j & 255;
  At[(size_t)w * NT + v] = (__bf16)S[j];            // At[w][v] = a[v][w]
}

// ---- adaptive adjacency: relu(nv1@nv2), mask zeros to -1e10, row softmax, transpose ----
__global__ void __launch_bounds__(256) adp_softmax_t(const float* __restrict__ nv1,
                                                     const float* __restrict__ nv2,
                                                     __bf16* __restrict__ At) {
  __shared__ float red[256];
  const int v = blockIdx.x, w = threadIdx.x;
  float a = 0.0f;
#pragma unroll
  for (int d = 0; d < 10; ++d) a = fmaf(nv1[v * 10 + d], nv2[d * NT + w], a);
  a = fmaxf(a, 0.0f);
  const float m = (a == 0.0f) ? -1e10f : a;
  red[w] = m; __syncthreads();
  for (int s = 128; s > 0; s >>= 1) { if (w < s) red[w] = fmaxf(red[w], red[w + s]); __syncthreads(); }
  const float mx = red[0]; __syncthreads();
  const float e = __expf(m - mx);
  red[w] = e; __syncthreads();
  for (int s = 128; s > 0; s >>= 1) { if (w < s) red[w] += red[w + s]; __syncthreads(); }
  const float p = e / red[0];
  At[(size_t)w * NT + v] = (__bf16)p;
}

__global__ void cvt_bf16(const float* __restrict__ src, __bf16* __restrict__ dst, int n) {
  const int i = blockIdx.x * 256 + threadIdx.x;
  if (i < n) dst[i] = (__bf16)src[i];
}

// ---- gated causal conv (kw=2), K = 32 channels; weights cached in LDS ----
// layout 0: X is [B][32][N][T] (standard), taps at t, t-dil  (stride 1)
// layout 1: X is [B][32][T][N] (time-major), taps at rows t, t-dil (stride NT)
// Output: bf16 time-major [B][32][T][N] (= segment slot 0)
__global__ void __launch_bounds__(256) conv_gated(const float* __restrict__ X,
    const float* __restrict__ wF, const float* __restrict__ bF,
    const float* __restrict__ wG, const float* __restrict__ bG,
    __bf16* __restrict__ Out, int layout, int dil) {
  __shared__ float swf[2048], swg[2048], sbf[32], sbg[32];
  for (int i = threadIdx.x; i < 2048; i += 256) { swf[i] = wF[i]; swg[i] = wG[i]; }
  if (threadIdx.x < 32) { sbf[threadIdx.x] = bF[threadIdx.x]; sbg[threadIdx.x] = bG[threadIdx.x]; }
  __syncthreads();

  const int j = blockIdx.x * 256 + threadIdx.x;     // B*N*T threads
  const int b = j >> 16, rem = j & (PLANE - 1);
  int t, outIdx, tap;
  if (layout == 0) { t = rem & 255; outIdx = ((rem & 255) << 8) | (rem >> 8); tap = -dil; }
  else             { t = rem >> 8;  outIdx = rem;                             tap = -dil * NT; }
  const bool valid = (t >= dil);
  const int tapo = valid ? tap : 0;                 // keep address in bounds

  const float* xb = X + (size_t)b * CH * PLANE + rem;
  float x0[CH], x1[CH];
#pragma unroll
  for (int i = 0; i < CH; ++i) {
    x0[i] = xb[(ptrdiff_t)i * PLANE];
    x1[i] = valid ? xb[(ptrdiff_t)i * PLANE + tapo] : 0.0f;
  }
  __bf16* ob = Out + (size_t)b * CH * PLANE + outIdx;
  for (int o = 0; o < CH; ++o) {
    float f = sbf[o], g = sbg[o];
    const float* wfo = &swf[o * 64];
    const float* wgo = &swg[o * 64];
#pragma unroll
    for (int i = 0; i < CH; ++i) {
      f = fmaf(wfo[2 * i], x1[i], f); f = fmaf(wfo[2 * i + 1], x0[i], f);
      g = fmaf(wgo[2 * i], x1[i], g); g = fmaf(wgo[2 * i + 1], x0[i], g);
    }
    const float h = f * (1.0f / (1.0f + __expf(-g)));
    ob[(size_t)o * PLANE] = (__bf16)h;
  }
}

// ---- diffusion hop as WMMA GEMM:  Out[t][w] = sum_v X[t][v] * a[v][w] = X(tn) @ A
// Xin/Out: bf16 [128 mats][T=256][N=256] (time-major per (b,c))
// AtB: bf16 [256][256] with AtB[w][v] = a[v][w] -> B-operand columns contiguous in K.
// Wave computes a 32x64 tile: 2 row-tiles share the 4 B-fragments
// -> 8 WMMAs per 12x16B loads per K-step. 512 blocks x 8 waves = 4096 jobs
//    (128 mats * 8 row-supertiles * 4 col-groups).
__global__ void __launch_bounds__(256) diffuse_wmma(const __bf16* __restrict__ Xin,
                                                    const __bf16* __restrict__ AtB,
                                                    __bf16* __restrict__ Out) {
  const int wave = threadIdx.x >> 5;
  const int lane = threadIdx.x & 31;
  const int hf = lane >> 4, l15 = lane & 15;
  const int job = blockIdx.x * 8 + wave;
  const int mat = job >> 5;
  const int ms  = (job >> 2) & 7;          // row supertile (32 rows)
  const int cg  = job & 3;                 // col group (64 cols)
  const int mr = ms * 32;
  const int nb = cg * 64;

  const __bf16* arow0 = Xin + (size_t)mat * PLANE + (size_t)(mr +      l15) * NT;
  const __bf16* arow1 = Xin + (size_t)mat * PLANE + (size_t)(mr + 16 + l15) * NT;
  const __bf16* bc0 = AtB + (size_t)(nb +  0 + l15) * NT;
  const __bf16* bc1 = AtB + (size_t)(nb + 16 + l15) * NT;
  const __bf16* bc2 = AtB + (size_t)(nb + 32 + l15) * NT;
  const __bf16* bc3 = AtB + (size_t)(nb + 48 + l15) * NT;

  v8f c00 = {}, c01 = {}, c02 = {}, c03 = {};
  v8f c10 = {}, c11 = {}, c12 = {}, c13 = {};
  for (int kb = 0; kb < NT; kb += 32) {
    v16bf af0 = load_a_frag(arow0 + kb, hf);
    v16bf af1 = load_a_frag(arow1 + kb, hf);
    v16bf b0 = load_b_frag(bc0 + kb, hf);
    v16bf b1 = load_b_frag(bc1 + kb, hf);
    v16bf b2 = load_b_frag(bc2 + kb, hf);
    v16bf b3 = load_b_frag(bc3 + kb, hf);
    c00 = __builtin_amdgcn_wmma_f32_16x16x32_bf16(false, af0, false, b0, (short)0, c00, false, false);
    c01 = __builtin_amdgcn_wmma_f32_16x16x32_bf16(false, af0, false, b1, (short)0, c01, false, false);
    c02 = __builtin_amdgcn_wmma_f32_16x16x32_bf16(false, af0, false, b2, (short)0, c02, false, false);
    c03 = __builtin_amdgcn_wmma_f32_16x16x32_bf16(false, af0, false, b3, (short)0, c03, false, false);
    c10 = __builtin_amdgcn_wmma_f32_16x16x32_bf16(false, af1, false, b0, (short)0, c10, false, false);
    c11 = __builtin_amdgcn_wmma_f32_16x16x32_bf16(false, af1, false, b1, (short)0, c11, false, false);
    c12 = __builtin_amdgcn_wmma_f32_16x16x32_bf16(false, af1, false, b2, (short)0, c12, false, false);
    c13 = __builtin_amdgcn_wmma_f32_16x16x32_bf16(false, af1, false, b3, (short)0, c13, false, false);
  }
  __bf16* obase = Out + (size_t)mat * PLANE;
  const int r0 = mr + hf * 8;
#pragma unroll
  for (int g = 0; g < 8; ++g) {
    __bf16* orow = obase + (size_t)(r0 + g) * NT + l15;
    orow[nb +  0] = (__bf16)c00[g];
    orow[nb + 16] = (__bf16)c01[g];
    orow[nb + 32] = (__bf16)c02[g];
    orow[nb + 48] = (__bf16)c03[g];
    __bf16* orow2 = orow + (size_t)16 * NT;
    orow2[nb +  0] = (__bf16)c10[g];
    orow2[nb + 16] = (__bf16)c11[g];
    orow2[nb + 32] = (__bf16)c12[g];
    orow2[nb + 48] = (__bf16)c13[g];
  }
}

// ---- 1x1 MLP as WMMA GEMM: out[o][j] = sum_{k<224} W[o][k] * SEG[k][j] + bias[o]
// SEG: bf16 [7][B][32][PLANE] (time-major); Wbf: bf16 [32][224].
// Block: b x 64 columns; 8 waves = 2 o-tiles x 4 col-tiles; B staged transposed in LDS.
__global__ void __launch_bounds__(256) mlp_wmma(const __bf16* __restrict__ SEG,
    const __bf16* __restrict__ Wbf, const float* __restrict__ bias,
    float* __restrict__ Out, int outlayout) {
  __shared__ __bf16 Bs[64][40];                      // [col][k], 80B rows (16B aligned)
  const int tid = threadIdx.x;
  const int b  = blockIdx.x >> 10;
  const int jb = (blockIdx.x & 1023) * 64;
  const int wave = tid >> 5, lane = tid & 31;
  const int hf = lane >> 4, l15 = lane & 15;
  const int mo = (wave & 1) * 16;
  const int nb = (wave >> 1) * 16;
  const int sr = tid >> 3;                           // staging: k row 0..31
  const int sg = (tid & 7) * 8;                      // staging: col start

  v8f acc = {};
  for (int kb = 0; kb < 224; kb += 32) {
    const int k = kb + sr;
    const int s = k >> 5, c = k & 31;
    const __bf16* rowp = SEG + (((size_t)s * BATCH + b) * CH + c) * PLANE + jb + sg;
    v8bf vv = *(const v8bf*)rowp;
#pragma unroll
    for (int e = 0; e < 8; ++e) Bs[sg + e][sr] = vv[e];   // transpose into LDS
    __syncthreads();
    v16bf af = load_a_frag(Wbf + (size_t)(mo + l15) * 224 + kb, hf);
    v16bf bfrag = load_b_frag(&Bs[nb + l15][0], hf);
    acc = __builtin_amdgcn_wmma_f32_16x16x32_bf16(false, af, false, bfrag, (short)0, acc, false, false);
    __syncthreads();
  }
  const int r0 = mo + hf * 8;
  const int col = jb + nb + l15;                     // j = t*256 + n
#pragma unroll
  for (int g = 0; g < 8; ++g) {
    const int o = r0 + g;
    const float val = acc[g] + bias[o];
    size_t addr;
    if (outlayout == 0)                              // standard [b][o][n][t]
      addr = (((size_t)b * CH + o) << 16) + ((size_t)(col & 255) << 8) + (size_t)(col >> 8);
    else                                             // time-major [b][o][t][n]
      addr = (((size_t)b * CH + o) << 16) + (size_t)col;
    Out[addr] = val;
  }
}

extern "C" void kernel_launch(void* const* d_in, const int* in_sizes, int n_in,
                              void* d_out, int out_size, void* d_ws, size_t ws_size,
                              hipStream_t stream) {
  const float* x    = (const float*)d_in[0];
  const float* nv1  = (const float*)d_in[1];
  const float* nv2  = (const float*)d_in[2];
  const float* sup0 = (const float*)d_in[3];
  const float* sup1 = (const float*)d_in[4];
  const float* w_f1 = (const float*)d_in[5];  const float* b_f1 = (const float*)d_in[6];
  const float* w_g1 = (const float*)d_in[7];  const float* b_g1 = (const float*)d_in[8];
  const float* w_f2 = (const float*)d_in[9];  const float* b_f2 = (const float*)d_in[10];
  const float* w_g2 = (const float*)d_in[11]; const float* b_g2 = (const float*)d_in[12];
  const float* w_m1 = (const float*)d_in[13]; const float* b_m1 = (const float*)d_in[14];
  const float* w_m2 = (const float*)d_in[15]; const float* b_m2 = (const float*)d_in[16];
  float* out = (float*)d_out;
  (void)in_sizes; (void)n_in; (void)out_size; (void)ws_size;

  // workspace carve
  char* ws = (char*)d_ws;
  size_t off = 0;
  auto carve = [&](size_t bytes) -> char* {
    char* p = ws + off;
    off += (bytes + 255) & ~(size_t)255;
    return p;
  };
  __bf16* At   = (__bf16*)carve((size_t)3 * PLANE * sizeof(__bf16));              // 3 transposed adjacencies
  __bf16* Wm   = (__bf16*)carve((size_t)2 * 32 * 224 * sizeof(__bf16));           // mlp weights bf16
  __bf16* seg  = (__bf16*)carve((size_t)7 * BATCH * CH * PLANE * sizeof(__bf16)); // 7 concat segments
  float*  hmid = (float*)carve((size_t)BATCH * CH * PLANE * sizeof(float));       // block-1 output (time-major)

  const size_t segPlane = (size_t)BATCH * CH * PLANE;   // elems per segment slot

  // adjacency + weight prep
  transpose_bf16<<<256, 256, 0, stream>>>(sup0, At + 0 * PLANE);
  transpose_bf16<<<256, 256, 0, stream>>>(sup1, At + 1 * PLANE);
  adp_softmax_t<<<256, 256, 0, stream>>>(nv1, nv2, At + 2 * PLANE);
  cvt_bf16<<<28, 256, 0, stream>>>(w_m1, Wm, 32 * 224);
  cvt_bf16<<<28, 256, 0, stream>>>(w_m2, Wm + 32 * 224, 32 * 224);

  // ---- block 1: dilation 1, input x in standard layout ----
  conv_gated<<<1024, 256, 0, stream>>>(x, w_f1, b_f1, w_g1, b_g1, seg, 0, 1);
  for (int ai = 0; ai < 3; ++ai) {
    diffuse_wmma<<<512, 256, 0, stream>>>(seg, At + (size_t)ai * PLANE,
                                          seg + (size_t)(1 + 2 * ai) * segPlane);
    diffuse_wmma<<<512, 256, 0, stream>>>(seg + (size_t)(1 + 2 * ai) * segPlane,
                                          At + (size_t)ai * PLANE,
                                          seg + (size_t)(2 + 2 * ai) * segPlane);
  }
  mlp_wmma<<<4096, 256, 0, stream>>>(seg, Wm, b_m1, hmid, 1);

  // ---- block 2: dilation 2, input hmid in time-major layout ----
  conv_gated<<<1024, 256, 0, stream>>>(hmid, w_f2, b_f2, w_g2, b_g2, seg, 1, 2);
  for (int ai = 0; ai < 3; ++ai) {
    diffuse_wmma<<<512, 256, 0, stream>>>(seg, At + (size_t)ai * PLANE,
                                          seg + (size_t)(1 + 2 * ai) * segPlane);
    diffuse_wmma<<<512, 256, 0, stream>>>(seg + (size_t)(1 + 2 * ai) * segPlane,
                                          At + (size_t)ai * PLANE,
                                          seg + (size_t)(2 + 2 * ai) * segPlane);
  }
  mlp_wmma<<<4096, 256, 0, stream>>>(seg, Wm + 32 * 224, b_m2, out, 0);
}
